// AtomAttentionEncoder_2963527434499
// MI455X (gfx1250) — matrix operand order repaired and checked by reading, not compiled, not run on hardware
//
#include <hip/hip_runtime.h>
#include <hip/hip_bf16.h>

// ---------------------------------------------------------------------------
// Types and WMMA helpers (gfx1250, wave32, v_wmma_f32_16x16x32_f16)
// ---------------------------------------------------------------------------
typedef __attribute__((ext_vector_type(16))) _Float16 v16h;
typedef __attribute__((ext_vector_type(8)))  float    v8f;

#define DEV __device__ __forceinline__

DEV v8f wmma(const v16h a, const v16h b, const v8f c) {
  return __builtin_amdgcn_wmma_f32_16x16x32_f16(false, a, false, b, (short)0, c,
                                                false, false);
}

DEV float wave_sum(float v) {
#pragma unroll
  for (int m = 16; m > 0; m >>= 1) v += __shfl_xor(v, m, 32);
  return v;
}
DEV float wave_max(float v) {
#pragma unroll
  for (int m = 16; m > 0; m >>= 1) v = fmaxf(v, __shfl_xor(v, m, 32));
  return v;
}
DEV float sigm(float x) { return 1.f / (1.f + __expf(-x)); }

// A fragment (16x32 f16) from LDS, row-major [M][ld], tile at (m0,k0).
// Per-lane halves are one contiguous 16-half run (k = 16*g + i) -> 2x ds_load_b128.
DEV v16h load_a_lds(const _Float16* X, int ld, int m0, int k0) {
  int lane = threadIdx.x & 31;
  int m = m0 + (lane & 15);
  int g = lane >> 4;
  v16h a;
#pragma unroll
  for (int i = 0; i < 16; ++i) {
    int k = g * 16 + i;
    a[i] = X[m * ld + k0 + k];
  }
  return a;
}
// B fragment from panel-transposed f16 weights in GLOBAL memory:
// layout [k/32][n][k%32]; a lane's 16 halves are contiguous -> 2x b128.
DEV v16h load_bp_g(const _Float16* __restrict__ W, int No, int k0, int n0) {
  int lane = threadIdx.x & 31;
  const _Float16* p = W + (size_t)(k0 >> 5) * 32 * No +
                      (size_t)(n0 + (lane & 15)) * 32 + (lane >> 4) * 16;
  v16h b;
#pragma unroll
  for (int i = 0; i < 16; ++i) b[i] = p[i];
  return b;
}
// Same but panel already staged in LDS ([n][32] layout).
DEV v16h load_bp_lds(const _Float16* sB, int n0) {
  int lane = threadIdx.x & 31;
  const _Float16* p = sB + (n0 + (lane & 15)) * 32 + (lane >> 4) * 16;
  v16h b;
#pragma unroll
  for (int i = 0; i < 16; ++i) b[i] = p[i];
  return b;
}
// B fragment from LDS f16 stored transposed [N][ld] (contiguous k run).
DEV v16h load_bT_lds(const _Float16* B, int ld, int k0, int n0) {
  int lane = threadIdx.x & 31;
  int n = n0 + (lane & 15);
  int g = lane >> 4;
  v16h b;
#pragma unroll
  for (int i = 0; i < 16; ++i) b[i] = B[n * ld + k0 + g * 16 + i];
  return b;
}
// K=16 padded variants (pair-MLP: K padded to 32 with zeros).
DEV v16h load_a_k16(const _Float16* X, int ld) {
  int lane = threadIdx.x & 31;
  int m = lane & 15;
  int g = lane >> 4;
  v16h a;
#pragma unroll
  for (int i = 0; i < 16; ++i) {
    int k = g * 16 + i;
    a[i] = (k < 16) ? X[m * ld + k] : (_Float16)0.f;
  }
  return a;
}
DEV v16h load_b_k16(const float* W) {  // 16x16 f32 in LDS
  int lane = threadIdx.x & 31;
  int n = lane & 15;
  int g = lane >> 4;
  v16h b;
#pragma unroll
  for (int i = 0; i < 16; ++i) {
    int k = g * 16 + i;
    b[i] = (k < 16) ? (_Float16)W[k * 16 + n] : (_Float16)0.f;
  }
  return b;
}

// ---------------------------------------------------------------------------
// K0: f32 -> f16 weight conversion into panel-transposed layout
//     dst[(k/32)*32*No + n*32 + (k%32)] = src[k*No + n]
// ---------------------------------------------------------------------------
__global__ void k_cvtw(const float* __restrict__ s, _Float16* __restrict__ d,
                       int Ni, int No) {
  int i = blockIdx.x * 256 + threadIdx.x;
  if (i < Ni * No) {
    int k = i / No, n = i % No;
    d[(size_t)(k >> 5) * 32 * No + (size_t)n * 32 + (k & 31)] = (_Float16)s[i];
  }
}

// ---------------------------------------------------------------------------
// K1: s_tok = LN(si; g,b) @ lin_s.W   [1024 x 384] -> [1024 x 128]
// ---------------------------------------------------------------------------
__global__ void k_stok(const float* __restrict__ si, const float* __restrict__ g,
                       const float* __restrict__ bb, const _Float16* __restrict__ WsH,
                       float* __restrict__ s_tok) {
  __shared__ _Float16 sA[16 * 384];
  int t0 = blockIdx.x * 16, tid = threadIdx.x;
  int wave = tid >> 5, lane = tid & 31;
#pragma unroll
  for (int rr = 0; rr < 2; ++rr) {
    int r = wave + rr * 8;
    const float* x = si + (size_t)(t0 + r) * 384;
    float vals[12], s = 0.f, s2 = 0.f;
#pragma unroll
    for (int u = 0; u < 12; ++u) {
      float t = x[lane * 12 + u];
      vals[u] = t; s += t; s2 += t * t;
    }
    s = wave_sum(s) * (1.f / 384.f);
    s2 = wave_sum(s2) * (1.f / 384.f);
    float inv = rsqrtf(fmaxf(s2 - s * s, 0.f) + 1e-5f);
#pragma unroll
    for (int u = 0; u < 12; ++u) {
      int c = lane * 12 + u;
      sA[r * 384 + c] = (_Float16)(((vals[u] - s) * inv) * g[c] + bb[c]);
    }
  }
  __syncthreads();
  int n0 = wave * 16;
  v8f c = {};
  for (int k0 = 0; k0 < 384; k0 += 32)
    c = wmma(load_a_lds(sA, 384, 0, k0), load_bp_g(WsH, 128, k0, n0), c);
  int n = n0 + (lane & 15), mb = (lane >> 4) * 8;
#pragma unroll
  for (int r = 0; r < 8; ++r) s_tok[(size_t)(t0 + mb + r) * 128 + n] = c[r];
}

// ---------------------------------------------------------------------------
// K2: atom embedder -> cl (ws + d_out), a = cl + rl@lin_r
// ---------------------------------------------------------------------------
__global__ void k_cl(const float* __restrict__ ref_pos, const float* __restrict__ charge,
                     const float* __restrict__ rmask, const float* __restrict__ elem,
                     const float* __restrict__ chars, const int* __restrict__ a2t,
                     const float* __restrict__ amask, const float* __restrict__ rl,
                     const float* __restrict__ s_tok,
                     const float* __restrict__ Wp, const float* __restrict__ bp,
                     const float* __restrict__ Wc, const float* __restrict__ bc,
                     const float* __restrict__ Wm, const float* __restrict__ bm,
                     const _Float16* __restrict__ WeH, const float* __restrict__ be,
                     const _Float16* __restrict__ WchH, const float* __restrict__ bch,
                     const float* __restrict__ Wr,
                     float* __restrict__ cl_ws, float* __restrict__ cl_out,
                     float* __restrict__ a_ws) {
  __shared__ _Float16 sE[16 * 128];
  __shared__ _Float16 sC[16 * 256];
  int m0 = blockIdx.x * 16, tid = threadIdx.x;
  for (int idx = tid; idx < 16 * 128; idx += 256)
    sE[idx] = (_Float16)elem[(size_t)m0 * 128 + idx];
  for (int idx = tid; idx < 16 * 256; idx += 256)
    sC[idx] = (_Float16)chars[(size_t)m0 * 256 + idx];
  __syncthreads();
  int wave = tid >> 5, lane = tid & 31, n0 = wave * 16;
  v8f c = {};
  for (int k0 = 0; k0 < 128; k0 += 32)
    c = wmma(load_a_lds(sE, 128, 0, k0), load_bp_g(WeH, 128, k0, n0), c);
  for (int k0 = 0; k0 < 256; k0 += 32)
    c = wmma(load_a_lds(sC, 256, 0, k0), load_bp_g(WchH, 128, k0, n0), c);
  int n = n0 + (lane & 15), mb = (lane >> 4) * 8;
#pragma unroll
  for (int r = 0; r < 8; ++r) {
    int m = m0 + mb + r;
    float y = c[r] + be[n] + bch[n] + bp[n] + bc[n] + bm[n];
    y += ref_pos[m * 3 + 0] * Wp[0 * 128 + n] + ref_pos[m * 3 + 1] * Wp[128 + n] +
         ref_pos[m * 3 + 2] * Wp[256 + n];
    float ch = charge[m];
    float as = logf(ch + sqrtf(ch * ch + 1.f));  // asinh
    y += as * Wc[n] + rmask[m] * Wm[n];
    y += s_tok[(size_t)a2t[m] * 128 + n] * amask[m];
    cl_ws[(size_t)m * 128 + n] = y;
    cl_out[(size_t)m * 128 + n] = y;
    a_ws[(size_t)m * 128 + n] = y + rl[m * 3 + 0] * Wr[n] +
                                rl[m * 3 + 1] * Wr[128 + n] +
                                rl[m * 3 + 2] * Wr[256 + n];
  }
}

// ---------------------------------------------------------------------------
// K3: row LayerNorm over 128 channels (no affine)
// ---------------------------------------------------------------------------
__global__ void k_ln128(const float* __restrict__ X, float* __restrict__ Yf,
                        _Float16* __restrict__ Yh) {
  int wave = threadIdx.x >> 5, lane = threadIdx.x & 31;
  size_t row = (size_t)blockIdx.x * 8 + wave;
  const float* x = X + row * 128;
  float v[4], s = 0.f, s2 = 0.f;
#pragma unroll
  for (int u = 0; u < 4; ++u) {
    v[u] = x[lane * 4 + u]; s += v[u]; s2 += v[u] * v[u];
  }
  s = wave_sum(s) * (1.f / 128.f);
  s2 = wave_sum(s2) * (1.f / 128.f);
  float inv = rsqrtf(fmaxf(s2 - s * s, 0.f) + 1e-5f);
#pragma unroll
  for (int u = 0; u < 4; ++u) {
    float y = (v[u] - s) * inv;
    if (Yf) Yf[row * 128 + lane * 4 + u] = y;
    if (Yh) Yh[row * 128 + lane * 4 + u] = (_Float16)y;
  }
}

// ---------------------------------------------------------------------------
// K4: pair features + deduped z gather + pair MLP -> plm (written to d_out)
// ---------------------------------------------------------------------------
__global__ void k_plm(const float* __restrict__ zij, const float* __restrict__ lnz_g,
                      const float* __restrict__ lnz_b, const _Float16* __restrict__ WzH,
                      const float* __restrict__ cl, const _Float16* __restrict__ WlH,
                      const float* __restrict__ bl, const _Float16* __restrict__ WmH,
                      const float* __restrict__ bm_, const float* __restrict__ ref_pos,
                      const int* __restrict__ uid, const float* __restrict__ amask,
                      const float* __restrict__ Woff, const float* __restrict__ boff,
                      const float* __restrict__ Winv, const float* __restrict__ binv,
                      const float* __restrict__ Wval, const float* __restrict__ bval,
                      const float* __restrict__ W1, const float* __restrict__ b1,
                      const float* __restrict__ W2, const float* __restrict__ b2,
                      const float* __restrict__ W3, float* __restrict__ plm_out) {
  __shared__ float z16[8 * 32 * 16];
  __shared__ float lq16[32 * 16];
  __shared__ float lm16[128 * 16];
  __shared__ __align__(16) unsigned char un[32768];
  __shared__ float sW[3 * 256];
  __shared__ float sB[2 * 16];
  __shared__ float sG[128], sBb[128];
  int b = blockIdx.x, tid = threadIdx.x, wave = tid >> 5, lane = tid & 31;
  int start = b * 32 - 48;
  if (tid < 256) { sW[tid] = W1[tid]; sW[256 + tid] = W2[tid]; sW[512 + tid] = W3[tid]; }
  if (tid < 16) { sB[tid] = b1[tid]; sB[16 + tid] = b2[tid]; }
  if (tid < 128) { sG[tid] = lnz_g[tid]; sBb[tid] = lnz_b[tid]; }
  _Float16* stage = (_Float16*)un + wave * (16 * 128);
  __syncthreads();

#pragma unroll 1
  for (int tt = 0; tt < 2; ++tt) {  // Phase A: deduped z16
    int t = wave + tt * 8;
    for (int m = 0; m < 16; ++m) {
      int p = t * 16 + m;
      int qi = p >> 5, sj = p & 31;
      int tq = b * 8 + qi;
      int ka = start + 4 * sj;
      ka = ka < 0 ? 0 : (ka > 4092 ? 4092 : ka);
      int tk = ka >> 2;
      const float* zr = zij + ((size_t)tq * 1024 + tk) * 128;
      float vv[4], s = 0.f, s2 = 0.f;
#pragma unroll
      for (int u = 0; u < 4; ++u) {
        float x = zr[lane * 4 + u]; vv[u] = x; s += x; s2 += x * x;
      }
      s = wave_sum(s) * (1.f / 128.f);
      s2 = wave_sum(s2) * (1.f / 128.f);
      float inv = rsqrtf(fmaxf(s2 - s * s, 0.f) + 1e-5f);
#pragma unroll
      for (int u = 0; u < 4; ++u) {
        int ci = lane * 4 + u;
        stage[m * 128 + ci] = (_Float16)(((vv[u] - s) * inv) * sG[ci] + sBb[ci]);
      }
    }
    __builtin_amdgcn_wave_barrier();
    v8f c = {};
    for (int k0 = 0; k0 < 128; k0 += 32)
      c = wmma(load_a_lds(stage, 128, 0, k0), load_bp_g(WzH, 16, k0, 0), c);
    int n = lane & 15, mb = (lane >> 4) * 8;
#pragma unroll
    for (int r = 0; r < 8; ++r) z16[(t * 16 + mb + r) * 16 + n] = c[r];
  }

#pragma unroll 1
  for (int pass = 0; pass < 2; ++pass) {  // Phase B: lq16 / lm16
    int tt = wave + pass * 8;
    if (tt < 10) {
      for (int m = 0; m < 16; ++m) {
        int arow;
        if (tt < 2) arow = b * 32 + tt * 16 + m;
        else {
          int j = (tt - 2) * 16 + m;
          int ka = start + j;
          arow = ka < 0 ? 0 : (ka > 4095 ? 4095 : ka);
        }
        const float* cr = cl + (size_t)arow * 128;
#pragma unroll
        for (int u = 0; u < 4; ++u) {
          int ci = lane * 4 + u;
          float x = cr[ci];
          stage[m * 128 + ci] = (_Float16)(x > 0.f ? x : 0.f);
        }
      }
      __builtin_amdgcn_wave_barrier();
      const _Float16* Wx = (tt < 2) ? WlH : WmH;
      const float* bx = (tt < 2) ? bl : bm_;
      v8f c = {};
      for (int k0 = 0; k0 < 128; k0 += 32)
        c = wmma(load_a_lds(stage, 128, 0, k0), load_bp_g(Wx, 16, k0, 0), c);
      int n = lane & 15, mb = (lane >> 4) * 8;
#pragma unroll
      for (int r = 0; r < 8; ++r) {
        int row = tt * 16 + mb + r;
        float y = c[r] + bx[n];
        if (tt < 2) lq16[row * 16 + n] = y;
        else lm16[(row - 32) * 16 + n] = y;
      }
    }
  }
  __syncthreads();

  _Float16* Abuf = (_Float16*)un;
  _Float16* Hbuf = (_Float16*)un + 256 * 16;
  float* Fbuf = (float*)(un + 2 * 256 * 16 * 2);
  for (int c0 = 0; c0 < 4096; c0 += 256) {  // Phase C
    int e = c0 + tid;
    int q = e >> 7, j = e & 127;
    int aq = b * 32 + q;
    int kr = start + j;
    float kvalid = (kr >= 0 && kr < 4096) ? 1.f : 0.f;
    int ak = kr < 0 ? 0 : (kr > 4095 ? 4095 : kr);
    float pm = amask[aq] * amask[ak] * kvalid;
    float dx = (ref_pos[aq * 3 + 0] - ref_pos[ak * 3 + 0]) * pm;
    float dy = (ref_pos[aq * 3 + 1] - ref_pos[ak * 3 + 1]) * pm;
    float dz = (ref_pos[aq * 3 + 2] - ref_pos[ak * 3 + 2]) * pm;
    float vlm = (uid[aq] == uid[ak]) ? pm : 0.f;
    float inv = 1.f / (1.f + dx * dx + dy * dy + dz * dz);
    float base[16];
#pragma unroll
    for (int n = 0; n < 16; ++n) {
      float t = (dx * Woff[n] + dy * Woff[16 + n] + dz * Woff[32 + n] + boff[n]) * vlm +
                (inv * Winv[n] + binv[n]) * vlm + (vlm * Wval[n] + bval[n]) * vlm +
                z16[((q >> 2) * 32 + (j >> 2)) * 16 + n] * pm +
                (lq16[q * 16 + n] + lm16[j * 16 + n]) * pm;
      base[n] = t;
      Abuf[tid * 16 + n] = (_Float16)(t > 0.f ? t : 0.f);
    }
    __syncthreads();
    for (int t = wave; t < 16; t += 8) {  // mlp1
      v8f c = wmma(load_a_k16(Abuf + t * 16 * 16, 16), load_b_k16(sW), (v8f){});
      int n = lane & 15, mb = (lane >> 4) * 8;
#pragma unroll
      for (int r = 0; r < 8; ++r) {
        float y = c[r] + sB[n];
        Hbuf[(t * 16 + mb + r) * 16 + n] = (_Float16)(y > 0.f ? y : 0.f);
      }
    }
    __syncthreads();
    for (int t = wave; t < 16; t += 8) {  // mlp2
      v8f c = wmma(load_a_k16(Hbuf + t * 16 * 16, 16), load_b_k16(sW + 256), (v8f){});
      int n = lane & 15, mb = (lane >> 4) * 8;
#pragma unroll
      for (int r = 0; r < 8; ++r) {
        float y = c[r] + sB[16 + n];
        Abuf[(t * 16 + mb + r) * 16 + n] = (_Float16)(y > 0.f ? y : 0.f);
      }
    }
    __syncthreads();
    for (int t = wave; t < 16; t += 8) {  // mlp3
      v8f c = wmma(load_a_k16(Abuf + t * 16 * 16, 16), load_b_k16(sW + 512), (v8f){});
      int n = lane & 15, mb = (lane >> 4) * 8;
#pragma unroll
      for (int r = 0; r < 8; ++r) Fbuf[(t * 16 + mb + r) * 16 + n] = c[r];
    }
    __syncthreads();
    size_t off = ((size_t)b * 4096 + e) * 16;
#pragma unroll
    for (int n = 0; n < 16; ++n)
      plm_out[off + n] = (base[n] + Fbuf[tid * 16 + n]) * pm;
    __syncthreads();
  }
}

// ---------------------------------------------------------------------------
// K5: generic fused GEMM with LDS-staged panel-transposed f16 B (k-outer).
// ---------------------------------------------------------------------------
__global__ void k_gemm(const _Float16* __restrict__ Xh, const _Float16* __restrict__ Wh,
                       const float* __restrict__ bias, float* __restrict__ Yf,
                       _Float16* __restrict__ Yh, const float* __restrict__ Gate,
                       float* __restrict__ Acc, int Ni, int No, int act) {
  __shared__ __align__(16) _Float16 sA[16 * 256];
  __shared__ __align__(16) _Float16 sB[32 * 384];
  int m0 = blockIdx.x * 16, tid = threadIdx.x;
  int wave = tid >> 5, lane = tid & 31;
  {  // A tile: rows contiguous -> 128-bit copy
    const uint4* src = (const uint4*)(Xh + (size_t)m0 * Ni);
    uint4* dst = (uint4*)sA;
    int cnt = (16 * Ni) >> 3;
    for (int idx = tid; idx < cnt; idx += 256) dst[idx] = src[idx];
  }
  int ntiles = No >> 4;
  int ntpw = (ntiles + 7) >> 3;  // 1..3 n-tiles per wave
  v8f acc[3];
#pragma unroll
  for (int t = 0; t < 3; ++t) acc[t] = (v8f){};
  for (int k0 = 0; k0 < Ni; k0 += 32) {
    __syncthreads();
    {  // panel p = k0/32 occupies a contiguous 32*No block -> b128 copy
      const uint4* src = (const uint4*)(Wh + (size_t)k0 * No);
      uint4* dst = (uint4*)sB;
      int cnt = (32 * No) >> 3;
      for (int idx = tid; idx < cnt; idx += 256) dst[idx] = src[idx];
    }
    if (k0 + 32 < Ni)  // prefetch next panel (global_prefetch_b8)
      __builtin_prefetch(Wh + (size_t)(k0 + 32) * No + (size_t)tid * 8, 0, 1);
    __syncthreads();
    v16h a = load_a_lds(sA, Ni, 0, k0);
#pragma unroll
    for (int t = 0; t < 3; ++t)
      if (t < ntpw)
        acc[t] = wmma(a, load_bp_lds(sB, (wave + t * 8) * 16), acc[t]);
  }
  int n = lane & 15, mb = (lane >> 4) * 8;
#pragma unroll
  for (int t = 0; t < 3; ++t) {
    if (t < ntpw) {
      int n0 = (wave + t * 8) * 16;
      float bv = bias ? bias[n0 + n] : 0.f;
#pragma unroll
      for (int r = 0; r < 8; ++r) {
        size_t m = (size_t)m0 + mb + r;
        float y = acc[t][r] + bv;
        if (act == 1) y = fmaxf(y, 0.f);
        else if (act == 2) y = sigm(y);
        if (Yf) Yf[m * No + n0 + n] = y;
        if (Yh) Yh[m * No + n0 + n] = (_Float16)y;
        if (Acc) Acc[m * No + n0 + n] += sigm(Gate[m * No + n0 + n]) * y;
      }
    }
  }
}

// ---------------------------------------------------------------------------
// K6: bias = LN(plm) @ Wb   (per pair entry, 16 -> 4)
// ---------------------------------------------------------------------------
__global__ void k_bias(const float* __restrict__ plm, const float* __restrict__ Wb,
                       float* __restrict__ biasb) {
  __shared__ float sWb[64];
  if (threadIdx.x < 64) sWb[threadIdx.x] = Wb[threadIdx.x];
  __syncthreads();
  size_t e = (size_t)blockIdx.x * 256 + threadIdx.x;
  float x[16], s = 0.f;
#pragma unroll
  for (int n = 0; n < 16; ++n) { x[n] = plm[e * 16 + n]; s += x[n]; }
  float mu = s * (1.f / 16.f), v = 0.f;
#pragma unroll
  for (int n = 0; n < 16; ++n) { float d = x[n] - mu; v += d * d; }
  float inv = rsqrtf(v * (1.f / 16.f) + 1e-5f);
#pragma unroll
  for (int hh = 0; hh < 4; ++hh) {
    float acc = 0.f;
#pragma unroll
    for (int n = 0; n < 16; ++n) acc += (x[n] - mu) * inv * sWb[n * 4 + hh];
    biasb[e * 4 + hh] = acc;
  }
}

// ---------------------------------------------------------------------------
// K7: block-windowed attention with pair bias, gate fused at output store.
//     V staged d-major so P@V B-fragments are contiguous (2x ds_load_b128).
// ---------------------------------------------------------------------------
__global__ void k_attn(const _Float16* __restrict__ qh, const _Float16* __restrict__ kh,
                       const _Float16* __restrict__ vh, const float* __restrict__ biasb,
                       const float* __restrict__ gate, const float* __restrict__ amask,
                       _Float16* __restrict__ oh) {
  __shared__ _Float16 kbuf[128 * 32];   // [j][d] (QK^T fragments contiguous)
  __shared__ _Float16 vbufT[32 * 128];  // [d][j] (PV fragments contiguous)
  __shared__ _Float16 qbuf[32 * 32], wbuf[32 * 128];
  __shared__ float lbuf[32 * 128];
  int b = blockIdx.x, tid = threadIdx.x, wave = tid >> 5, lane = tid & 31;
  int start = b * 32 - 48;
  const float inv_s = 0.17677669529663687f;
#pragma unroll 1
  for (int h = 0; h < 4; ++h) {
    for (int idx = tid; idx < 32 * 32; idx += 256) {
      int m = idx >> 5, d = idx & 31;
      qbuf[idx] = qh[(size_t)(b * 32 + m) * 128 + h * 32 + d];
    }
    for (int idx = tid; idx < 128 * 32; idx += 256) {
      int j = idx >> 5, d = idx & 31;
      int ka = start + j;
      ka = ka < 0 ? 0 : (ka > 4095 ? 4095 : ka);
      kbuf[idx] = kh[(size_t)ka * 128 + h * 32 + d];
      vbufT[d * 128 + j] = vh[(size_t)ka * 128 + h * 32 + d];
    }
    __syncthreads();
    for (int job = wave; job < 16; job += 8) {
      int qt = job >> 3, kt = job & 7;
      v8f c = wmma(load_a_lds(qbuf, 32, qt * 16, 0),
                   load_bT_lds(kbuf, 32, 0, kt * 16), (v8f){});
      int n = lane & 15, mb = (lane >> 4) * 8;
#pragma unroll
      for (int r = 0; r < 8; ++r) {
        int qq = qt * 16 + mb + r, kk = kt * 16 + n;
        int krr = start + kk;
        float kv = (krr >= 0 && krr < 4096) ? 1.f : 0.f;
        int ka = krr < 0 ? 0 : (krr > 4095 ? 4095 : krr);
        float pm = amask[b * 32 + qq] * amask[ka] * kv;
        float l = c[r] * inv_s + biasb[((size_t)(b * 32 + qq) * 128 + kk) * 4 + h];
        lbuf[qq * 128 + kk] = (pm > 0.f) ? l : -1e9f;
      }
    }
    __syncthreads();
    for (int q = wave; q < 32; q += 8) {
      float vals[4], vmax = -1e30f;
#pragma unroll
      for (int u = 0; u < 4; ++u) {
        vals[u] = lbuf[q * 128 + lane * 4 + u];
        vmax = fmaxf(vmax, vals[u]);
      }
      vmax = wave_max(vmax);
      float s = 0.f;
#pragma unroll
      for (int u = 0; u < 4; ++u) { vals[u] = __expf(vals[u] - vmax); s += vals[u]; }
      s = wave_sum(s);
      float rs = 1.f / s;
#pragma unroll
      for (int u = 0; u < 4; ++u)
        wbuf[q * 128 + lane * 4 + u] = (_Float16)(vals[u] * rs);
    }
    __syncthreads();
    if (wave < 4) {
      int qt = wave >> 1, dt = wave & 1;
      v8f c = {};
      for (int k0 = 0; k0 < 128; k0 += 32)
        c = wmma(load_a_lds(wbuf, 128, qt * 16, k0),
                 load_bT_lds(vbufT, 128, k0, dt * 16), c);
      int n = lane & 15, mb = (lane >> 4) * 8;
#pragma unroll
      for (int r = 0; r < 8; ++r) {
        size_t aq = (size_t)b * 32 + qt * 16 + mb + r;
        int col = h * 32 + dt * 16 + n;
        float g = gate[aq * 128 + col];
        oh[aq * 128 + col] = (_Float16)(c[r] * sigm(g));
      }
    }
    __syncthreads();
  }
}

// ---------------------------------------------------------------------------
// Small elementwise / aggregation kernels
// ---------------------------------------------------------------------------
__global__ void k_ac(const float* __restrict__ g, const float* __restrict__ sh,
                     const float* __restrict__ lna, _Float16* __restrict__ ach, int n) {
  int i = blockIdx.x * 256 + threadIdx.x;
  if (i < n) ach[i] = (_Float16)(sigm(g[i]) * lna[i] + sh[i]);
}
__global__ void k_silu(const float* __restrict__ u1, const float* __restrict__ u2,
                       _Float16* __restrict__ uh, int n) {
  int i = blockIdx.x * 256 + threadIdx.x;
  if (i < n) { float a = u1[i]; uh[i] = (_Float16)((a * sigm(a)) * u2[i]); }
}
__global__ void k_qlout(const float* __restrict__ a, const float* __restrict__ amask,
                        float* __restrict__ qlf, _Float16* __restrict__ qlh, int n) {
  int i = blockIdx.x * 256 + threadIdx.x;
  if (i < n) {
    float y = a[i] * amask[i >> 7];
    qlf[i] = y; qlh[i] = (_Float16)y;
  }
}
__global__ void k_scatter(const float* __restrict__ qfeat, const int* __restrict__ a2t,
                          const float* __restrict__ amask, float* __restrict__ num, int n) {
  int i = blockIdx.x * 256 + threadIdx.x;
  if (i < n) {
    int row = i / 384, c = i % 384;
    atomicAdd(&num[(size_t)a2t[row] * 384 + c], qfeat[i] * amask[row]);
  }
}
__global__ void k_den(const int* __restrict__ a2t, const float* __restrict__ amask,
                      float* __restrict__ den, int n) {
  int i = blockIdx.x * 256 + threadIdx.x;
  if (i < n) atomicAdd(&den[a2t[i]], amask[i]);
}
__global__ void k_ai(const float* __restrict__ num, const float* __restrict__ den,
                     const float* __restrict__ tmask, float* __restrict__ ai, int n) {
  int i = blockIdx.x * 256 + threadIdx.x;
  if (i < n) {
    int t = i / 384;
    ai[i] = num[i] / fmaxf(den[t], 1.f) * tmask[t];
  }
}

// ---------------------------------------------------------------------------
// Host launcher (param index map as documented in round 1)
// ---------------------------------------------------------------------------
extern "C" void kernel_launch(void* const* d_in, const int* in_sizes, int n_in,
                              void* d_out, int out_size, void* d_ws, size_t ws_size,
                              hipStream_t stream) {
  (void)in_sizes; (void)n_in; (void)out_size; (void)ws_size;
  const float* ref_pos = (const float*)d_in[0];
  const float* charge = (const float*)d_in[1];
  const float* rmask = (const float*)d_in[2];
  const float* elem = (const float*)d_in[3];
  const float* chars = (const float*)d_in[4];
  const int* uid = (const int*)d_in[5];
  const float* amask = (const float*)d_in[6];
  const float* tmask = (const float*)d_in[7];
  const int* a2t = (const int*)d_in[8];
  const float* rl = (const float*)d_in[9];
  const float* si = (const float*)d_in[10];
  const float* zij = (const float*)d_in[11];
  auto P = [&](int i) { return (const float*)d_in[i]; };

  char* ws = (char*)d_ws;
  size_t off = 0;
  auto alloc = [&](size_t bytes) {
    void* p = ws + off;
    off = (off + bytes + 255) & ~(size_t)255;
    return p;
  };
  float* cl_ws = (float*)alloc(4096 * 128 * 4);
  float* a_ws = (float*)alloc(4096 * 128 * 4);
  float* s_tok = (float*)alloc(1024 * 128 * 4);
  _Float16* s_n_h = (_Float16*)alloc(4096 * 128 * 2);
  float* ln_a = (float*)alloc(4096 * 128 * 4);
  float* g_ada = (float*)alloc(4096 * 128 * 4);
  float* sh_ada = (float*)alloc(4096 * 128 * 4);
  _Float16* ac_h = (_Float16*)alloc(4096 * 128 * 2);
  _Float16* q_h = (_Float16*)alloc(4096 * 128 * 2);
  _Float16* kk_h = (_Float16*)alloc(4096 * 128 * 2);
  _Float16* v_h = (_Float16*)alloc(4096 * 128 * 2);
  float* gate_f = (float*)alloc(4096 * 128 * 4);
  float* wog_f = (float*)alloc(4096 * 128 * 4);
  _Float16* o_h = (_Float16*)alloc(4096 * 128 * 2);
  float* biasb = (float*)alloc((size_t)128 * 32 * 128 * 4 * 4);
  float* u1 = (float*)alloc(4096 * 256 * 4);
  float* u2 = (float*)alloc(4096 * 256 * 4);
  _Float16* u_h = (_Float16*)alloc(4096 * 256 * 2);
  _Float16* ql_h = (_Float16*)alloc(4096 * 128 * 2);
  float* qfeat = (float*)alloc(4096 * 384 * 4);
  float* num = (float*)alloc(1024 * 384 * 4);
  float* den = (float*)alloc(1024 * 4);

  // panel-transposed f16 weight copies: per layer
  // [0]=Wgate [1]=Wk [2]=Wo [3]=Wv [4]=Wog.W [5]=Wq.W [6]=ada_g.W [7]=ada_sh.W
  // [8]=W1 [9]=W2 [10]=W3 [11]=tWog.W [12]=tada_g.W [13]=tada_sh.W
  const int wNi[14] = {128, 128, 128, 128, 128, 128, 128, 128,
                       128, 128, 256, 128, 128, 128};
  const int wNo[14] = {128, 128, 128, 128, 128, 128, 128, 128,
                       256, 256, 128, 128, 128, 128};
  const int widx[14] = {1, 2, 3, 8, 4, 6, 9, 11, 12, 13, 14, 15, 17, 19};
  _Float16* wh[3][14];
  for (int l = 0; l < 3; ++l)
    for (int k = 0; k < 14; ++k)
      wh[l][k] = (_Float16*)alloc((size_t)wNi[k] * wNo[k] * 2);
  _Float16* whq = (_Float16*)alloc(128 * 384 * 2);  // lin_q
  _Float16* WsH = (_Float16*)alloc(384 * 128 * 2);  // lin_s
  _Float16* WeH = (_Float16*)alloc(128 * 128 * 2);  // ref_element
  _Float16* WchH = (_Float16*)alloc(256 * 128 * 2); // ref_chars
  _Float16* WzH = (_Float16*)alloc(128 * 16 * 2);   // lin_z
  _Float16* WlH = (_Float16*)alloc(128 * 16 * 2);   // lin_l
  _Float16* WmH = (_Float16*)alloc(128 * 16 * 2);   // lin_m

  float* out = (float*)d_out;
  float* out_ai = out;
  float* out_ql = out + 393216;
  float* out_cl = out + 917504;
  float* out_plm = out + 1441792;

  for (int l = 0; l < 3; ++l) {
    int L = 14 + 20 * l;
    for (int k = 0; k < 14; ++k) {
      int n = wNi[k] * wNo[k];
      k_cvtw<<<(n + 255) / 256, 256, 0, stream>>>(P(L + widx[k]), wh[l][k],
                                                  wNi[k], wNo[k]);
    }
  }
  k_cvtw<<<(128 * 384 + 255) / 256, 256, 0, stream>>>(P(78), whq, 128, 384);
  k_cvtw<<<(384 * 128 + 255) / 256, 256, 0, stream>>>(P(80), WsH, 384, 128);
  k_cvtw<<<(128 * 128 + 255) / 256, 256, 0, stream>>>(P(95), WeH, 128, 128);
  k_cvtw<<<(256 * 128 + 255) / 256, 256, 0, stream>>>(P(93), WchH, 256, 128);
  k_cvtw<<<(128 * 16 + 255) / 256, 256, 0, stream>>>(P(81), WzH, 128, 16);
  k_cvtw<<<(128 * 16 + 255) / 256, 256, 0, stream>>>(P(74), WlH, 128, 16);
  k_cvtw<<<(128 * 16 + 255) / 256, 256, 0, stream>>>(P(76), WmH, 128, 16);

  k_stok<<<64, 256, 0, stream>>>(si, P(83), P(82), WsH, s_tok);
  k_cl<<<256, 256, 0, stream>>>(ref_pos, charge, rmask, elem, chars, a2t, amask, rl,
                                s_tok, P(101), P(102), P(91), P(92), P(97), P(98),
                                WeH, P(96), WchH, P(94), P(79), cl_ws, out_cl, a_ws);
  k_ln128<<<512, 256, 0, stream>>>(cl_ws, nullptr, s_n_h);
  k_plm<<<128, 256, 0, stream>>>(zij, P(85), P(84), WzH, cl_ws, WlH, P(75), WmH,
                                 P(77), ref_pos, uid, amask, P(99), P(100), P(12),
                                 P(13), P(103), P(104), P(86), P(87), P(88), P(89),
                                 P(90), out_plm);

  const int NE = 4096 * 128;
  for (int l = 0; l < 3; ++l) {
    int L = 14 + 20 * l;
    _Float16** w = &wh[l][0];
    k_ln128<<<512, 256, 0, stream>>>(a_ws, ln_a, nullptr);
    k_gemm<<<256, 256, 0, stream>>>(s_n_h, w[6], P(L + 10), g_ada, nullptr, nullptr,
                                    nullptr, 128, 128, 0);
    k_gemm<<<256, 256, 0, stream>>>(s_n_h, w[7], nullptr, sh_ada, nullptr, nullptr,
                                    nullptr, 128, 128, 0);
    k_ac<<<NE / 256, 256, 0, stream>>>(g_ada, sh_ada, ln_a, ac_h, NE);
    k_gemm<<<256, 256, 0, stream>>>(ac_h, w[5], P(L + 7), nullptr, q_h, nullptr,
                                    nullptr, 128, 128, 0);
    k_gemm<<<256, 256, 0, stream>>>(ac_h, w[1], nullptr, nullptr, kk_h, nullptr,
                                    nullptr, 128, 128, 0);
    k_gemm<<<256, 256, 0, stream>>>(ac_h, w[3], nullptr, nullptr, v_h, nullptr,
                                    nullptr, 128, 128, 0);
    k_gemm<<<256, 256, 0, stream>>>(ac_h, w[0], nullptr, gate_f, nullptr, nullptr,
                                    nullptr, 128, 128, 0);
    k_bias<<<2048, 256, 0, stream>>>(out_plm, P(L + 0), biasb);
    k_attn<<<128, 256, 0, stream>>>(q_h, kk_h, v_h, biasb, gate_f, amask, o_h);
    k_gemm<<<256, 256, 0, stream>>>(s_n_h, w[4], P(L + 5), wog_f, nullptr, nullptr,
                                    nullptr, 128, 128, 0);
    k_gemm<<<256, 256, 0, stream>>>(o_h, w[2], nullptr, nullptr, nullptr, wog_f,
                                    a_ws, 128, 128, 0);
    k_ln128<<<512, 256, 0, stream>>>(a_ws, ln_a, nullptr);
    k_gemm<<<256, 256, 0, stream>>>(s_n_h, w[12], P(L + 18), g_ada, nullptr, nullptr,
                                    nullptr, 128, 128, 0);
    k_gemm<<<256, 256, 0, stream>>>(s_n_h, w[13], nullptr, sh_ada, nullptr, nullptr,
                                    nullptr, 128, 128, 0);
    k_ac<<<NE / 256, 256, 0, stream>>>(g_ada, sh_ada, ln_a, ac_h, NE);
    k_gemm<<<256, 256, 0, stream>>>(ac_h, w[8], nullptr, u1, nullptr, nullptr,
                                    nullptr, 128, 256, 0);
    k_gemm<<<256, 256, 0, stream>>>(ac_h, w[9], nullptr, u2, nullptr, nullptr,
                                    nullptr, 128, 256, 0);
    k_silu<<<(4096 * 256) / 256, 256, 0, stream>>>(u1, u2, u_h, 4096 * 256);
    k_gemm<<<256, 256, 0, stream>>>(s_n_h, w[11], P(L + 16), wog_f, nullptr, nullptr,
                                    nullptr, 128, 128, 0);
    k_gemm<<<256, 256, 0, stream>>>(u_h, w[10], nullptr, nullptr, nullptr, wog_f,
                                    a_ws, 256, 128, 0);
  }

  k_qlout<<<NE / 256, 256, 0, stream>>>(a_ws, amask, out_ql, ql_h, NE);
  k_gemm<<<256, 256, 0, stream>>>(ql_h, whq, nullptr, qfeat, nullptr, nullptr,
                                  nullptr, 128, 384, 1);
  hipMemsetAsync(num, 0, 1024 * 384 * 4, stream);
  hipMemsetAsync(den, 0, 1024 * 4, stream);
  k_scatter<<<(4096 * 384) / 256, 256, 0, stream>>>(qfeat, a2t, amask, num, 4096 * 384);
  k_den<<<16, 256, 0, stream>>>(a2t, amask, den, 4096);
  k_ai<<<(1024 * 384) / 256, 256, 0, stream>>>(num, den, tmask, out_ai, 1024 * 384);
}
